// IntraModalSoftCrossEntropyLoss_7095285973555
// MI455X (gfx1250) — compile-verified, matrix operand order checked by reading, not run on hardware
//
#include <hip/hip_runtime.h>
#include <hip/hip_bf16.h>
#include <math.h>

// ---------------------------------------------------------------------------
// Fused IntraModalSoftCrossEntropyLoss for MI455X (gfx1250, wave32, WMMA).
//   cos   = (F/||F||) @ (G/||G||)^T          N=M=4096, D=1024
//   inter = L @ Lc^T                          C=80
//   loss  = mean( -(P log(Q+e) + (1-P) log(1-Q+e)) ),  P=2*inter/(sumL+sumLc+e)
// f32 emulated via bf16 hi/lo split: x*y ~= hi*hi + hi*lo + lo*hi  (3 WMMAs)
// ---------------------------------------------------------------------------

typedef __attribute__((ext_vector_type(16))) __bf16 v16bf;
typedef __attribute__((ext_vector_type(8)))  __bf16 v8bf;
typedef __attribute__((ext_vector_type(4)))  __bf16 v4bf;
typedef __attribute__((ext_vector_type(8)))  float  v8f;

#define N_ROWS 4096
#define M_ROWS 4096
#define DDIM   1024
#define CDIM   80

#define BT   128   // block tile edge (M and N)
#define KB   32    // K step (bf16 WMMA depth)
#define LDT  40    // LDS row stride in bf16 elems: 80B (16B aligned, conflict-breaking)

// ---- load a 128 x 32 f32 tile -> bf16 hi/lo planes in LDS (scaled, padded) ----
__device__ __forceinline__ void load_tile(const float* __restrict__ src, int ldk, int klim,
                                          int rowBase, int k0,
                                          const float* __restrict__ scale,
                                          __bf16* sHi, __bf16* sLo, int tid)
{
#pragma unroll
  for (int i = 0; i < 4; ++i) {
    int s  = tid + i * 256;          // 1024 float4 slots: 128 rows x 8 chunks
    int r  = s >> 3;
    int c4 = (s & 7) << 2;
    int gr = rowBase + r;
    int gc = k0 + c4;
    float4 v = make_float4(0.f, 0.f, 0.f, 0.f);
    if (gc + 3 < klim) {
      const float* p = src + (size_t)gr * ldk + gc;
      v = *(const float4*)p;
      __builtin_prefetch(p + KB, 0, 1);   // next K-step row chunk -> global_prefetch_b8
    }
    float sc = scale ? scale[gr] : 1.0f;
    float x0 = v.x * sc, x1 = v.y * sc, x2 = v.z * sc, x3 = v.w * sc;
    __bf16 h0 = (__bf16)x0, h1 = (__bf16)x1, h2 = (__bf16)x2, h3 = (__bf16)x3;
    __bf16 l0 = (__bf16)(x0 - (float)h0);
    __bf16 l1 = (__bf16)(x1 - (float)h1);
    __bf16 l2 = (__bf16)(x2 - (float)h2);
    __bf16 l3 = (__bf16)(x3 - (float)h3);
    v4bf hv; hv.x = h0; hv.y = h1; hv.z = h2; hv.w = h3;
    v4bf lv; lv.x = l0; lv.y = l1; lv.z = l2; lv.w = l3;
    *(v4bf*)(sHi + r * LDT + c4) = hv;
    *(v4bf*)(sLo + r * LDT + c4) = lv;
  }
}

// ---- build one 16x32 bf16 A/B fragment per ISA 7.12.2 lane layout ----
// lanes 0-15: K = 0..7 (vgpr0-3) and 16..23 (vgpr4-7); lanes 16-31: +8.
__device__ __forceinline__ v16bf frag(const __bf16* plane, int rc, int koff)
{
  v8bf p0 = *(const v8bf*)(plane + rc * LDT + koff);
  v8bf p1 = *(const v8bf*)(plane + rc * LDT + 16 + koff);
  return __builtin_shufflevector(p0, p1, 0,1,2,3,4,5,6,7,8,9,10,11,12,13,14,15);
}

// ---- one K-step of the 128x128 tile: 24 WMMAs per wave ----
__device__ __forceinline__ void mma_step(const __bf16* sAhi, const __bf16* sAlo,
                                         const __bf16* sBhi, const __bf16* sBlo,
                                         v8f* acc, int wm, int wn, int lr, int koff)
{
  v16bf ah[2], al[2];
#pragma unroll
  for (int tm = 0; tm < 2; ++tm) {
    int rr = wm * 32 + tm * 16 + lr;
    ah[tm] = frag(sAhi, rr, koff);
    al[tm] = frag(sAlo, rr, koff);
  }
#pragma unroll
  for (int tn = 0; tn < 4; ++tn) {
    int cc = wn * 64 + tn * 16 + lr;
    v16bf bh = frag(sBhi, cc, koff);
    v16bf bl = frag(sBlo, cc, koff);
#pragma unroll
    for (int tm = 0; tm < 2; ++tm) {
      int idx = tm * 4 + tn;
      acc[idx] = __builtin_amdgcn_wmma_f32_16x16x32_bf16(false, ah[tm], false, bh,
                                                         (short)0, acc[idx], false, false);
      acc[idx] = __builtin_amdgcn_wmma_f32_16x16x32_bf16(false, ah[tm], false, bl,
                                                         (short)0, acc[idx], false, false);
      acc[idx] = __builtin_amdgcn_wmma_f32_16x16x32_bf16(false, al[tm], false, bh,
                                                         (short)0, acc[idx], false, false);
    }
  }
}

// ---------------------------------------------------------------------------
__global__ __launch_bounds__(256)
void prep_norms(const float* __restrict__ F, const float* __restrict__ G,
                float* __restrict__ invF, float* __restrict__ invC)
{
  __shared__ float red[256];
  int row = blockIdx.x;
  const float* src = (row < N_ROWS) ? F : G;
  int r = (row < N_ROWS) ? row : row - N_ROWS;
  int tid = threadIdx.x;
  float s = 0.f;
#pragma unroll
  for (int i = 0; i < 4; ++i) {
    float x = src[(size_t)r * DDIM + tid + i * 256];
    s += x * x;
  }
  red[tid] = s;
  __syncthreads();
  for (int o = 128; o > 0; o >>= 1) {
    if (tid < o) red[tid] += red[tid + o];
    __syncthreads();
  }
  if (tid == 0) {
    float inv = 1.f / fmaxf(sqrtf(red[0]), 1e-8f);
    if (row < N_ROWS) invF[r] = inv; else invC[r] = inv;
  }
}

__global__ void prep_sums(const float* __restrict__ L, const float* __restrict__ Lc,
                          float* __restrict__ sumL, float* __restrict__ sumC,
                          double* __restrict__ acc)
{
  int t = blockIdx.x * blockDim.x + threadIdx.x;
  if (t == 0) *acc = 0.0;                 // re-zero every launch (graph replay safe)
  if (t < 2 * N_ROWS) {
    const float* src = (t < N_ROWS) ? L : Lc;
    int r = (t < N_ROWS) ? t : t - N_ROWS;
    float s = 0.f;
    for (int j = 0; j < CDIM; ++j) s += src[(size_t)r * CDIM + j];
    if (t < N_ROWS) sumL[r] = s; else sumC[r] = s;
  }
}

__global__ __launch_bounds__(256)
void fused_loss_kernel(const float* __restrict__ F, const float* __restrict__ G,
                       const float* __restrict__ L, const float* __restrict__ Lc,
                       const float* __restrict__ invF, const float* __restrict__ invC,
                       const float* __restrict__ sumL, const float* __restrict__ sumC,
                       double* __restrict__ accOut)
{
  __shared__ __bf16 sAhi[BT * LDT], sAlo[BT * LDT];
  __shared__ __bf16 sBhi[BT * LDT], sBlo[BT * LDT];
  __shared__ float sred[256];

  const int tid  = threadIdx.x;
  const int lane = tid & 31;
  const int lr   = lane & 15;
  const int koff = (lane & 16) ? 8 : 0;
  const int wave = tid >> 5;       // 8 waves
  const int wm   = wave & 3;       // 4 row groups of 32 rows
  const int wn   = wave >> 2;      // 2 col groups of 64 cols
  const int rowBase = blockIdx.y * BT;
  const int colBase = blockIdx.x * BT;

  v8f accC[8], accP[8];
  v8f zero = {0.f, 0.f, 0.f, 0.f, 0.f, 0.f, 0.f, 0.f};
#pragma unroll
  for (int i = 0; i < 8; ++i) { accC[i] = zero; accP[i] = zero; }

  // ---- cosine GEMM: K = 1024, normalization fused into LDS load ----
  for (int k0 = 0; k0 < DDIM; k0 += KB) {
    load_tile(F, DDIM, DDIM, rowBase, k0, invF, sAhi, sAlo, tid);
    load_tile(G, DDIM, DDIM, colBase, k0, invC, sBhi, sBlo, tid);
    __syncthreads();
    mma_step(sAhi, sAlo, sBhi, sBlo, accC, wm, wn, lr, koff);
    __syncthreads();
  }

  // ---- label-overlap GEMM: K = 80, zero-padded to 96 ----
  for (int k0 = 0; k0 < 96; k0 += KB) {
    load_tile(L,  CDIM, CDIM, rowBase, k0, nullptr, sAhi, sAlo, tid);
    load_tile(Lc, CDIM, CDIM, colBase, k0, nullptr, sBhi, sBlo, tid);
    __syncthreads();
    mma_step(sAhi, sAlo, sBhi, sBlo, accP, wm, wn, lr, koff);
    __syncthreads();
  }

  // ---- epilogue: P, sigmoid, logs, partial mean ----
  float local = 0.f;
#pragma unroll
  for (int tm = 0; tm < 2; ++tm) {
#pragma unroll
    for (int tn = 0; tn < 4; ++tn) {
      int idx = tm * 4 + tn;
      int col = colBase + wn * 64 + tn * 16 + lr;
      float sc = sumC[col];
#pragma unroll
      for (int r = 0; r < 8; ++r) {
        int row = rowBase + wm * 32 + tm * 16 + r + ((lane & 16) ? 8 : 0);
        float c     = accC[idx][r];
        float inter = accP[idx][r];
        float un = sumL[row] + sc + 1e-8f;
        float Pv = 2.f * inter / un;
        float q  = 1.f / (1.f + __expf(-c));
        local -= Pv * __logf(q + 1e-8f) + (1.f - Pv) * __logf(1.f - q + 1e-8f);
      }
    }
  }
  sred[tid] = local;
  __syncthreads();
  for (int o = 128; o > 0; o >>= 1) {
    if (tid < o) sred[tid] += sred[tid + o];
    __syncthreads();
  }
  if (tid == 0) atomicAdd(accOut, (double)sred[0]);
}

__global__ void finalize_kernel(const double* __restrict__ acc, float* __restrict__ out)
{
  out[0] = (float)(acc[0] / ((double)N_ROWS * (double)M_ROWS));
}

// ---------------------------------------------------------------------------
extern "C" void kernel_launch(void* const* d_in, const int* in_sizes, int n_in,
                              void* d_out, int out_size, void* d_ws, size_t ws_size,
                              hipStream_t stream)
{
  const float* F  = (const float*)d_in[0];   // features      (4096 x 1024)
  const float* G  = (const float*)d_in[1];   // comp_features (4096 x 1024)
  const float* L  = (const float*)d_in[2];   // labels        (4096 x 80)
  const float* Lc = (const float*)d_in[3];   // comp_labels   (4096 x 80)
  float* out = (float*)d_out;

  // workspace: [ f64 accumulator | invF | invC | sumL | sumC ]  (~64 KB)
  double* acc  = (double*)d_ws;
  float*  invF = (float*)d_ws + 4;           // 16B offset, keeps acc aligned
  float*  invC = invF + N_ROWS;
  float*  sumL = invC + M_ROWS;
  float*  sumC = sumL + N_ROWS;

  prep_norms<<<dim3(N_ROWS + M_ROWS), 256, 0, stream>>>(F, G, invF, invC);
  prep_sums<<<dim3((2 * N_ROWS + 255) / 256), 256, 0, stream>>>(L, Lc, sumL, sumC, acc);
  fused_loss_kernel<<<dim3(M_ROWS / BT, N_ROWS / BT), 256, 0, stream>>>(
      F, G, L, Lc, invF, invC, sumL, sumC, acc);
  finalize_kernel<<<1, 1, 0, stream>>>(acc, out);
}
